// AdditiveAttention_66099546685553
// MI455X (gfx1250) — matrix operand order repaired and checked by reading, compiled, not run on hardware
//
#include <hip/hip_runtime.h>
#include <hip/hip_bf16.h>
#include <math.h>

typedef __attribute__((ext_vector_type(16))) _Float16 v16h;
typedef __attribute__((ext_vector_type(8)))  _Float16 v8h;
typedef __attribute__((ext_vector_type(8)))  float    v8f;

#define BB   16
#define CC   256
#define KCC  32
#define NN   2304     // 48*48
#define NT16 144      // NN/16
#define NT32 72       // NN/32
#define ROWS 32       // i-rows per attention block (2 sub-tiles of 16)

// ---------------------------------------------------------------------------
// Kernel A: projections q = Wq x + bq, k = Wk x + bk, v = Wv x + bv,
// converted to f16 and scattered directly into WMMA fragment layouts.
//   qfrag[b][it][lane][e]     : A-matrix 16x32 f16 layout (M = i, K = kc)
//   kfrag[b][jt][lane][e]     : B-matrix 32x16 f16 layout (K = kc, N = j)
//   vfrag[b][js][ct][lane][e] : B-matrix 32x16 f16 layout (K = j-local, N = c)
// ---------------------------------------------------------------------------
__global__ __launch_bounds__(256) void proj_kernel(
    const float* __restrict__ x,
    const float* __restrict__ Wq, const float* __restrict__ bq,
    const float* __restrict__ Wk, const float* __restrict__ bk,
    const float* __restrict__ Wv, const float* __restrict__ bv,
    _Float16* __restrict__ qfrag,
    _Float16* __restrict__ kfrag,
    _Float16* __restrict__ vfrag)
{
    __shared__ float xs[CC][32];          // 32 KB: x[b, :, n0..n0+31]

    const int nt = blockIdx.x;            // 0..71
    const int b  = blockIdx.y;            // 0..15
    const int n0 = nt * 32;
    const int t  = threadIdx.x;

    // stage x tile into LDS (coalesced float4 loads)
    const float* xb = x + ((size_t)b * CC) * NN + n0;
    #pragma unroll
    for (int kk = 0; kk < 8; ++kk) {
        int f  = t + kk * 256;            // float4 id 0..2047
        int c  = f >> 3;
        int j4 = (f & 7) * 4;
        float4 val = *(const float4*)(xb + (size_t)c * NN + j4);
        *(float4*)(&xs[c][j4]) = val;
    }
    __syncthreads();

    // ---- V rows: thread t owns output channel co = t, 32 columns ----
    {
        const int co = t;
        float acc[32];
        const float bvv = bv[co];
        #pragma unroll
        for (int j = 0; j < 32; ++j) acc[j] = bvv;
        const float* wrow = Wv + (size_t)co * CC;
        for (int c = 0; c < CC; ++c) {
            const float w = wrow[c];
            const float4* xr = (const float4*)(xs[c]);
            #pragma unroll
            for (int q4 = 0; q4 < 8; ++q4) {
                float4 xv = xr[q4];
                acc[q4 * 4 + 0] = fmaf(w, xv.x, acc[q4 * 4 + 0]);
                acc[q4 * 4 + 1] = fmaf(w, xv.y, acc[q4 * 4 + 1]);
                acc[q4 * 4 + 2] = fmaf(w, xv.z, acc[q4 * 4 + 2]);
                acc[q4 * 4 + 3] = fmaf(w, xv.w, acc[q4 * 4 + 3]);
            }
        }
        // write two contiguous 16-half B-fragment rows (lane = g*16 + (co&15))
        v16h o0, o1;
        #pragma unroll
        for (int j = 0; j < 16; ++j) {
            o0[j] = (_Float16)acc[j];
            o1[j] = (_Float16)acc[j + 16];
        }
        size_t basev = ((((size_t)b * NT32 + nt) * 16 + (co >> 4)) * 32 + (co & 15)) * 16;
        *(v16h*)(vfrag + basev)            = o0;   // g=0: j-local 0..15
        *(v16h*)(vfrag + basev + 16 * 16)  = o1;   // g=1: j-local 16..31
    }

    // ---- Q/K rows: threads 0..63 ----
    if (t < 64) {
        const bool isq = (t < 32);
        const int  r   = t & 31;          // kc row
        const float* wrow = (isq ? Wq : Wk) + (size_t)r * CC;
        const float  bias = (isq ? bq : bk)[r];
        float acc[32];
        #pragma unroll
        for (int j = 0; j < 32; ++j) acc[j] = bias;
        for (int c = 0; c < CC; ++c) {
            const float w = wrow[c];
            const float4* xr = (const float4*)(xs[c]);
            #pragma unroll
            for (int q4 = 0; q4 < 8; ++q4) {
                float4 xv = xr[q4];
                acc[q4 * 4 + 0] = fmaf(w, xv.x, acc[q4 * 4 + 0]);
                acc[q4 * 4 + 1] = fmaf(w, xv.y, acc[q4 * 4 + 1]);
                acc[q4 * 4 + 2] = fmaf(w, xv.z, acc[q4 * 4 + 2]);
                acc[q4 * 4 + 3] = fmaf(w, xv.w, acc[q4 * 4 + 3]);
            }
        }
        if (isq) {
            // A-layout: K = 8g + (e&7) + 16*(e>>3)  => g = bit3(K), e = (bit4<<3)|(K&7)
            const int gg = (r >> 3) & 1;
            const int e  = ((r >> 4) << 3) | (r & 7);
            #pragma unroll
            for (int j = 0; j < 32; ++j) {
                const int n    = n0 + j;
                const int tile = n >> 4;
                const int m    = n & 15;
                qfrag[(((size_t)b * NT16 + tile) * 32 + (gg * 16 + m)) * 16 + e] =
                    (_Float16)acc[j];
            }
        } else {
            // B-layout: K = 8g + (e&7) + 16*(e>>3)  => g = bit3(K), e = (bit4<<3)|(K&7)
            const int gg = (r >> 3) & 1;
            const int e  = ((r >> 4) << 3) | (r & 7);
            #pragma unroll
            for (int j = 0; j < 32; ++j) {
                const int n    = n0 + j;
                const int tile = n >> 4;
                const int nc   = n & 15;
                kfrag[(((size_t)b * NT16 + tile) * 32 + (gg * 16 + nc)) * 16 + e] =
                    (_Float16)acc[j];
            }
        }
    }
}

// ---------------------------------------------------------------------------
// Kernel B: attention for one (b, 32-row i-tile). 8 waves / block.
// LDS: S (32 x 2304 f16 = 144 KB) + softmax scratch; ~149 KB total
//      -> 2 workgroups per 320 KB WGP.
// Pass1: 2 score sub-tiles per j-tile via v_wmma_f32_16x16x32_f16 (K = KC)
// Pass2: two-pass softmax, all S access as 8-half vectors (ds_*_b128)
// Pass3: O += P @ V^T ; each V fragment reused by 2 row sub-tiles
//        (4 WMMAs per 4 ds_b128 + 4 global_b128)
// Epilogue: out = gamma*O + x (fused, vectorized)
// ---------------------------------------------------------------------------
__global__ __launch_bounds__(256) void attn_kernel(
    const float* __restrict__ x,
    const float* __restrict__ gamma,
    const _Float16* __restrict__ qfrag,
    const _Float16* __restrict__ kfrag,
    const _Float16* __restrict__ vfrag,
    float* __restrict__ out)
{
    extern __shared__ char smem[];
    _Float16* S    = (_Float16*)smem;                     // 32*2304 halves = 144 KB
    float*    red  = (float*)(smem + ROWS * NN * 2);      // 32x8 partials
    float*    rowm = red + ROWS * 8;                      // 32 row maxima
    float*    rowl = rowm + ROWS;                         // 32 inverse row sums

    const int it0  = blockIdx.x;        // 32-row tile, 0..71
    const int b    = blockIdx.y;        // batch
    const int lane = threadIdx.x & 31;
    const int wave = threadIdx.x >> 5;  // 0..7
    const float scale = 0.17677669529663687f;  // 1/sqrt(KC)

    // Q A-fragments for the two 16-row sub-tiles (persistent in registers)
    const v16h qa0 = *(const v16h*)(qfrag + (((size_t)b * NT16 + 2 * it0    ) * 32 + lane) * 16);
    const v16h qa1 = *(const v16h*)(qfrag + (((size_t)b * NT16 + 2 * it0 + 1) * 32 + lane) * 16);

    // ---- Pass 1: scores (2 WMMAs per j-tile, shared K fragment) ----
    for (int tt = 0; tt < NT16 / 8; ++tt) {
        const int jt = wave + tt * 8;
        const v16h kb = *(const v16h*)(kfrag + (((size_t)b * NT16 + jt) * 32 + lane) * 16);
        v8f c0 = {}, c1 = {};
        c0 = __builtin_amdgcn_wmma_f32_16x16x32_f16(false, qa0, false, kb,
                                                    (short)0, c0, false, false);
        c1 = __builtin_amdgcn_wmma_f32_16x16x32_f16(false, qa1, false, kb,
                                                    (short)0, c1, false, false);
        const int jcol = jt * 16 + (lane & 15);
        const int g    = lane >> 4;
        #pragma unroll
        for (int r = 0; r < 8; ++r) {
            S[(size_t)(r + 8 * g)      * NN + jcol] = (_Float16)(c0[r] * scale);
            S[(size_t)(r + 8 * g + 16) * NN + jcol] = (_Float16)(c1[r] * scale);
        }
    }
    __syncthreads();

    // ---- Pass 2: softmax; thread owns 8-half groups (row = tid>>3) ----
    {
        const int row  = threadIdx.x >> 3;   // 0..31
        const int cpos = threadIdx.x & 7;
        const _Float16* Sr = S + (size_t)row * NN;
        float m = -3.0e38f;
        for (int j8 = cpos; j8 < NN / 8; j8 += 8) {
            const v8h sv = *(const v8h*)(Sr + j8 * 8);
            #pragma unroll
            for (int k = 0; k < 8; ++k) m = fmaxf(m, (float)sv[k]);
        }
        red[row * 8 + cpos] = m;
        __syncthreads();
        if (cpos == 0) {
            float mm = red[row * 8];
            #pragma unroll
            for (int i = 1; i < 8; ++i) mm = fmaxf(mm, red[row * 8 + i]);
            rowm[row] = mm;
        }
        __syncthreads();
        const float mrow = rowm[row];
        float s = 0.0f;
        for (int j8 = cpos; j8 < NN / 8; j8 += 8) {
            const v8h sv = *(const v8h*)(Sr + j8 * 8);
            #pragma unroll
            for (int k = 0; k < 8; ++k) s += __expf((float)sv[k] - mrow);
        }
        red[row * 8 + cpos] = s;
        __syncthreads();
        if (cpos == 0) {
            float ss = 0.0f;
            #pragma unroll
            for (int i = 0; i < 8; ++i) ss += red[row * 8 + i];
            rowl[row] = 1.0f / ss;
        }
        __syncthreads();
        const float linv = rowl[row];
        _Float16* Sw = S + (size_t)row * NN;
        for (int j8 = cpos; j8 < NN / 8; j8 += 8) {   // in-place P (owner-exclusive)
            v8h sv = *(const v8h*)(Sw + j8 * 8);
            v8h pv;
            #pragma unroll
            for (int k = 0; k < 8; ++k)
                pv[k] = (_Float16)(__expf((float)sv[k] - mrow) * linv);
            *(v8h*)(Sw + j8 * 8) = pv;
        }
    }
    __syncthreads();

    // ---- Pass 3: O += P @ V^T ; wave owns C-tiles 2*wave, 2*wave+1 ----
    v8f acc00 = {}, acc01 = {}, acc10 = {}, acc11 = {};
    const int m   = lane & 15;
    const int g   = lane >> 4;
    const int ct0 = wave * 2;
    for (int js = 0; js < NT32; ++js) {
        // P A-fragments for both row sub-tiles: two 16B runs each -> ds_load_b128
        const _Float16* pp0 = S + (size_t)m        * NN + js * 32 + g * 16;
        const _Float16* pp1 = S + (size_t)(m + 16) * NN + js * 32 + g * 16;
        const v8h lo0 = *(const v8h*)pp0;
        const v8h hi0 = *(const v8h*)(pp0 + 8);
        const v8h lo1 = *(const v8h*)pp1;
        const v8h hi1 = *(const v8h*)(pp1 + 8);
        v16h a0, a1;
        #pragma unroll
        for (int i = 0; i < 8; ++i) {
            a0[i] = lo0[i]; a0[i + 8] = hi0[i];
            a1[i] = lo1[i]; a1[i + 8] = hi1[i];
        }

        const size_t vb = (((size_t)b * NT32 + js) * 16);
        const v16h b0 = *(const v16h*)(vfrag + ((vb + ct0    ) * 32 + lane) * 16);
        const v16h b1 = *(const v16h*)(vfrag + ((vb + ct0 + 1) * 32 + lane) * 16);

        acc00 = __builtin_amdgcn_wmma_f32_16x16x32_f16(false, a0, false, b0,
                                                       (short)0, acc00, false, false);
        acc01 = __builtin_amdgcn_wmma_f32_16x16x32_f16(false, a0, false, b1,
                                                       (short)0, acc01, false, false);
        acc10 = __builtin_amdgcn_wmma_f32_16x16x32_f16(false, a1, false, b0,
                                                       (short)0, acc10, false, false);
        acc11 = __builtin_amdgcn_wmma_f32_16x16x32_f16(false, a1, false, b1,
                                                       (short)0, acc11, false, false);
    }

    // ---- Epilogue: out = gamma*O + x (D rows land on 8 consecutive n) ----
    const float gm = gamma[0];
    const v8f* accs[4] = { &acc00, &acc01, &acc10, &acc11 };
    #pragma unroll
    for (int s = 0; s < 2; ++s) {
        #pragma unroll
        for (int tcol = 0; tcol < 2; ++tcol) {
            const int c = (ct0 + tcol) * 16 + m;
            const size_t base = ((size_t)b * CC + c) * NN + it0 * 32 + s * 16 + 8 * g;
            const v8f* av = accs[s * 2 + tcol];
            float4 x0 = *(const float4*)(x + base);
            float4 x1 = *(const float4*)(x + base + 4);
            float4 o0, o1;
            o0.x = fmaf(gm, (*av)[0], x0.x); o0.y = fmaf(gm, (*av)[1], x0.y);
            o0.z = fmaf(gm, (*av)[2], x0.z); o0.w = fmaf(gm, (*av)[3], x0.w);
            o1.x = fmaf(gm, (*av)[4], x1.x); o1.y = fmaf(gm, (*av)[5], x1.y);
            o1.z = fmaf(gm, (*av)[6], x1.z); o1.w = fmaf(gm, (*av)[7], x1.w);
            *(float4*)(out + base)     = o0;
            *(float4*)(out + base + 4) = o1;
        }
    }
}

// ---------------------------------------------------------------------------
extern "C" void kernel_launch(void* const* d_in, const int* in_sizes, int n_in,
                              void* d_out, int out_size, void* d_ws, size_t ws_size,
                              hipStream_t stream) {
    (void)in_sizes; (void)n_in; (void)out_size; (void)ws_size;
    const float* x     = (const float*)d_in[0];
    const float* Wq    = (const float*)d_in[1];
    const float* bq    = (const float*)d_in[2];
    const float* Wk    = (const float*)d_in[3];
    const float* bk    = (const float*)d_in[4];
    const float* Wv    = (const float*)d_in[5];
    const float* bv    = (const float*)d_in[6];
    const float* gamma = (const float*)d_in[7];
    float* out = (float*)d_out;

    // workspace: q/k A/B fragments + v B fragments (f16), ~23.6 MB total
    const size_t qk_elems = (size_t)BB * NT16 * 32 * 16;          // 1,179,648 halves each
    _Float16* qfrag = (_Float16*)d_ws;
    _Float16* kfrag = qfrag + qk_elems;
    _Float16* vfrag = kfrag + qk_elems;                            // BB*NT32*16*32*16

    dim3 gridA(NT32, BB);
    proj_kernel<<<gridA, 256, 0, stream>>>(x, Wq, bq, Wk, bk, Wv, bv,
                                           qfrag, kfrag, vfrag);

    dim3 gridB(NN / ROWS, BB);
    const size_t lds = (size_t)ROWS * NN * 2
                     + (ROWS * 8 + ROWS + ROWS) * sizeof(float);   // ~149 KB
    attn_kernel<<<gridB, 256, lds, stream>>>(x, gamma, qfrag, kfrag, vfrag, out);
}
